// Attention_32839319945876
// MI455X (gfx1250) — matrix operand order repaired and checked by reading, not compiled, not run on hardware
//
#include <hip/hip_runtime.h>
#include <hip/hip_bf16.h>

#define T_SEQ 1024
#define C_DIM 768
#define NH    12
#define NBR   4
#define DH    64
#define HT    48
#define KRET  12
#define EPSF  1.1920929e-07f

typedef __bf16 v16bf __attribute__((ext_vector_type(16)));
typedef float  v8f   __attribute__((ext_vector_type(8)));
typedef unsigned int v4u __attribute__((ext_vector_type(4)));

union BF16x16 { v16bf v; unsigned short u[16]; v4u q[2]; };

__device__ __forceinline__ unsigned short f2bf(float f) {
    unsigned int u = __float_as_uint(f);
    return (unsigned short)((u + 0x7FFFu + ((u >> 16) & 1u)) >> 16);
}
__device__ __forceinline__ unsigned int pack2bf(float lo, float hi) {
    unsigned int a = __float_as_uint(lo), b = __float_as_uint(hi);
    unsigned int ra = (a + 0x7FFFu + ((a >> 16) & 1u)) >> 16;
    unsigned int rb = (b + 0x7FFFu + ((b >> 16) & 1u)) & 0xFFFF0000u;
    return rb | ra;
}

// A fragment: 16x32 bf16 (MxK) from row-major bf16. A lane holds 16
// contiguous k = kb..kb+15 (matching B) -> two contiguous 16B runs.
__device__ __forceinline__ v16bf load_a_frag_bf(const __bf16* A, int lda, int lane) {
    int m  = lane & 15;
    int kb = (lane < 16) ? 0 : 16;
    const __bf16* p = A + (size_t)m * lda + kb;
    BF16x16 r;
    r.q[0] = *(const v4u*)p;          // k = kb .. kb+7
    r.q[1] = *(const v4u*)(p + 8);    // k = kb+8 .. kb+15
    return r.v;
}
// B fragment: 32x16 bf16 (KxN) from N-major bf16 (Wt[n][k]). Per lane: 16
// contiguous k values -> two 16B loads.
__device__ __forceinline__ v16bf load_b_frag_bf(const __bf16* Wt, int ldk, int lane) {
    int n  = lane & 15;
    int kb = (lane < 16) ? 0 : 16;
    const __bf16* p = Wt + (size_t)n * ldk + kb;
    BF16x16 r;
    r.q[0] = *(const v4u*)p;
    r.q[1] = *(const v4u*)(p + 8);
    return r.v;
}

__device__ __forceinline__ v8f wmma_bf16(v16bf a, v16bf b, v8f c) {
    return __builtin_amdgcn_wmma_f32_16x16x32_bf16(false, a, false, b, (short)0, c, false, false);
}

// ------------------------------------------------------------ prologue: cvt
__global__ __launch_bounds__(256) void cvt_bf16_kernel(
    const float* __restrict__ in, unsigned int* __restrict__ out, int n2) {
    int i = blockIdx.x * blockDim.x + threadIdx.x;
    if (i < n2) out[i] = pack2bf(in[2 * i], in[2 * i + 1]);
}

// out[z][n][k] = bf16(in[z][k][n])  (K, N multiples of 32)
__global__ __launch_bounds__(256) void transpose_bf16_kernel(
    const float* __restrict__ in, unsigned short* __restrict__ out, int K, int N) {
    __shared__ float tile[32][33];
    size_t base = (size_t)blockIdx.z * K * N;
    int k0 = blockIdx.x * 32, n0 = blockIdx.y * 32;
    int tx = threadIdx.x & 31, ty = threadIdx.x >> 5;
    for (int i = ty; i < 32; i += 8)
        tile[i][tx] = in[base + (size_t)(k0 + i) * N + n0 + tx];
    __syncthreads();
    for (int i = ty; i < 32; i += 8)
        out[base + (size_t)(n0 + i) * K + k0 + tx] = f2bf(tile[tx][i]);
}

// ------------------------------------- GEMM + bias, 16x64 per-wave (4 accum)
__global__ __launch_bounds__(256) void gemm_bias_kernel(
    const __bf16* __restrict__ A, const __bf16* __restrict__ Wt,
    const float* __restrict__ bias, float* __restrict__ out,
    int M, int K, int N) {
    int lane  = threadIdx.x & 31;
    int wid   = blockIdx.x * (blockDim.x >> 5) + (threadIdx.x >> 5);
    int nblk  = N >> 6;                      // 64-wide column blocks
    int total = (M >> 4) * nblk;
    if (wid >= total) return;
    int tm = wid / nblk, tn = wid % nblk;
    v8f cacc[4] = {};
    const __bf16* Ap = A + (size_t)tm * 16 * K;
    const __bf16* Wp = Wt + (size_t)tn * 64 * K;
    for (int k0 = 0; k0 < K; k0 += 32) {
        __builtin_prefetch(Ap + k0 + 64, 0, 3);
        v16bf a = load_a_frag_bf(Ap + k0, K, lane);
#pragma unroll
        for (int j = 0; j < 4; ++j) {
            v16bf b = load_b_frag_bf(Wp + (size_t)j * 16 * K + k0, K, lane);
            cacc[j] = wmma_bf16(a, b, cacc[j]);
        }
    }
    int mh = (lane >> 4) & 1;
    int nn = lane & 15;
#pragma unroll
    for (int j = 0; j < 4; ++j) {
        int n = tn * 64 + j * 16 + nn;
        float bv = bias[n];
#pragma unroll
        for (int v = 0; v < 8; ++v) {
            int m = tm * 16 + v + 8 * mh;
            out[(size_t)m * N + n] = cacc[j][v] + bv;
        }
    }
}

// ---------------------------------------------------------- head prep kernels
__device__ __forceinline__ void wedge_rope(float* x, const float* swA,
                                           const float* __restrict__ wbias,
                                           int h, int t, float* out) {
    float y[DH];
#pragma unroll 4
    for (int e = 0; e < DH; ++e) {
        float acc = x[e] * (1.0f + wbias[h * DH + e]);
        for (int d = 0; d < DH; ++d)
            acc += x[d] * (swA[d * DH + e] - swA[e * DH + d]);
        y[e] = acc;
    }
#pragma unroll
    for (int i = 0; i < 32; ++i) {
        float invf = __expf(-(2.0f * i / 64.0f) * 9.210340371976184f); // ln(1e4)
        float sn, cs;
        __sincosf((float)t * invf, &sn, &cs);
        out[i]      = y[2 * i] * cs - y[2 * i + 1] * sn;
        out[32 + i] = y[2 * i] * sn + y[2 * i + 1] * cs;
    }
}

__global__ __launch_bounds__(256) void qprep_kernel(
    const float* __restrict__ qlin, const float* __restrict__ wA,
    const float* __restrict__ wbias, unsigned short* __restrict__ qr) {
    __shared__ float swA[DH * DH];
    int h = blockIdx.x;
    int t = blockIdx.y * blockDim.x + threadIdx.x;
    for (int i = threadIdx.x; i < DH * DH; i += blockDim.x) swA[i] = wA[i];
    __syncthreads();
    const float* src = qlin + (size_t)t * (NBR * C_DIM) + h * DH;
    float x[DH];
    float ss = 0.f;
#pragma unroll
    for (int d = 0; d < DH; ++d) { x[d] = src[d]; ss += x[d] * x[d]; }
    float inv = rsqrtf(ss * (1.0f / DH) + EPSF);
#pragma unroll
    for (int d = 0; d < DH; ++d) x[d] *= inv;
    float out[DH];
    wedge_rope(x, swA, wbias, h, t, out);
    unsigned int* dst = (unsigned int*)(qr + ((size_t)h * T_SEQ + t) * DH);
#pragma unroll
    for (int d = 0; d < DH / 2; ++d) dst[d] = pack2bf(out[2 * d], out[2 * d + 1]);
}

__global__ __launch_bounds__(256) void kprep_kernel(
    const float* __restrict__ klin, const float* __restrict__ wA,
    const float* __restrict__ wbias, unsigned short* __restrict__ kr,
    float* __restrict__ kbaseh) {
    __shared__ float swA[DH * DH];
    int h  = blockIdx.x;
    int hb = h % NH;
    int t  = blockIdx.y * blockDim.x + threadIdx.x;
    for (int i = threadIdx.x; i < DH * DH; i += blockDim.x) swA[i] = wA[i];
    __syncthreads();
    const float* src = klin + (size_t)t * C_DIM + hb * DH;
    float x[DH];
#pragma unroll
    for (int d = 0; d < DH; ++d) x[d] = src[d];
    if (h < NH) {
        float* kb = kbaseh + ((size_t)hb * T_SEQ + t) * DH;
#pragma unroll
        for (int d = 0; d < DH; ++d) kb[d] = x[d];
    }
    float out[DH];
    wedge_rope(x, swA, wbias, h, t, out);
    unsigned int* dst = (unsigned int*)(kr + ((size_t)h * T_SEQ + t) * DH);
#pragma unroll
    for (int d = 0; d < DH / 2; ++d) dst[d] = pack2bf(out[2 * d], out[2 * d + 1]);
}

// ------------------------------------------- scores + softmax + top-k + marker
#define SCLD 1024
__global__ __launch_bounds__(512) void attn_topk_kernel(
    const __bf16* __restrict__ qr, const __bf16* __restrict__ kr,
    const float* __restrict__ kbaseh, const float* __restrict__ sink,
    unsigned short* __restrict__ marker, float* __restrict__ psink) {
    __shared__ float sc[16 * SCLD];   // 64 KB score panel (of 320 KB WGP LDS)
    int qtile = blockIdx.x;
    int h     = blockIdx.y;
    int lane  = threadIdx.x & 31;
    int w     = threadIdx.x >> 5;     // 16 waves
    int mh    = (lane >> 4) & 1;
    int nn    = lane & 15;
    const float scale = 0.125f;       // 64^-0.5

    // phase 1: WMMA score panel into LDS (key tiles striped over waves);
    // kr's [t][d] layout is already N-major for the B operand.
    const __bf16* Aq = qr + ((size_t)h * T_SEQ + qtile * 16) * DH;
    v16bf a0 = load_a_frag_bf(Aq, DH, lane);
    v16bf a1 = load_a_frag_bf(Aq + 32, DH, lane);
    for (int kt = w; kt <= qtile; kt += 16) {
        const __bf16* Kp = kr + ((size_t)h * T_SEQ + kt * 16) * DH;
        v16bf b0 = load_b_frag_bf(Kp, DH, lane);
        v16bf b1 = load_b_frag_bf(Kp + 32, DH, lane);
        v8f c = {};
        c = wmma_bf16(a0, b0, c);
        c = wmma_bf16(a1, b1, c);
#pragma unroll
        for (int v = 0; v < 8; ++v) {
            int m  = v + 8 * mh;
            int tq = qtile * 16 + m;
            int sk = kt * 16 + nn;
            sc[m * SCLD + sk] = (sk <= tq) ? c[v] * scale : -1e30f;
        }
    }
    __syncthreads();

    // phase 2: one wave per query row
    int m = w;
    int t = qtile * 16 + m;
    const float* row = sc + m * SCLD;
    float topv[KRET]; int topi[KRET];
#pragma unroll
    for (int j = 0; j < KRET; ++j) { topv[j] = -1e30f; topi[j] = -1; }
    float rmax = -1e30f;
    for (int s = lane; s <= t; s += 32) {
        float v = row[s];
        rmax = fmaxf(rmax, v);
        if (v > topv[KRET - 1]) {
            topv[KRET - 1] = v; topi[KRET - 1] = s;
#pragma unroll
            for (int j = KRET - 1; j > 0; --j) {
                if (topv[j] > topv[j - 1]) {
                    float tv = topv[j]; topv[j] = topv[j - 1]; topv[j - 1] = tv;
                    int   ti = topi[j]; topi[j] = topi[j - 1]; topi[j - 1] = ti;
                }
            }
        }
    }
#pragma unroll
    for (int off = 16; off > 0; off >>= 1)
        rmax = fmaxf(rmax, __shfl_xor(rmax, off, 32));
    float snk  = sink[h];
    float Mx   = fmaxf(rmax, snk);
    float sexp = 0.f;
    for (int s = lane; s <= t; s += 32) sexp += __expf(row[s] - Mx);
#pragma unroll
    for (int off = 16; off > 0; off >>= 1) sexp += __shfl_xor(sexp, off, 32);
    float esink = __expf(snk - Mx);
    float denom = sexp + esink;
    float pse   = esink / denom;

    // merge per-lane sorted lists: 12 rounds of wave arg-max
    int p = 0;
    float wts[KRET]; int gidx[KRET];
    float wsum = 0.f;
#pragma unroll
    for (int j = 0; j < KRET; ++j) {
        float f  = (p < KRET) ? topv[p] : -1e30f;
        int   ci = (p < KRET) ? topi[p] : -1;
        float bm = f; int bl = lane;
#pragma unroll
        for (int off = 16; off > 0; off >>= 1) {
            float of = __shfl_xor(bm, off, 32);
            int   ol = __shfl_xor(bl, off, 32);
            if (of > bm || (of == bm && ol < bl)) { bm = of; bl = ol; }
        }
        int bi = __shfl(ci, bl, 32);
        if (lane == bl) p++;
        float e = (bm > -1e29f) ? __expf(bm - Mx) / denom : 0.f;
        wts[j] = e; gidx[j] = bi; wsum += e;
    }
    float winv = 1.0f / (wsum + 1e-9f);
    int hb = h % NH;
    float acc0 = 0.f, acc1 = 0.f;
#pragma unroll
    for (int j = 0; j < KRET; ++j) {
        int idx = gidx[j];
        if (idx >= 0) {
            const float* kp = kbaseh + ((size_t)hb * T_SEQ + idx) * DH;
            float wj = wts[j] * winv;
            acc0 += wj * kp[lane];
            acc1 += wj * kp[lane + 32];
        }
    }
    unsigned short* mp = marker + ((size_t)h * T_SEQ + t) * DH;
    mp[lane]      = f2bf(acc0);
    mp[lane + 32] = f2bf(acc1);
    if (lane == 0) psink[(size_t)h * T_SEQ + t] = pse;
}

// ---------------------------------------------------- fused MLP + sink -> ctx
__global__ __launch_bounds__(256) void mlp_ctx_kernel(
    const __bf16* __restrict__ marker, const __bf16* __restrict__ fct,
    const float* __restrict__ fc_b, const __bf16* __restrict__ projt,
    const float* __restrict__ proj_b, const float* __restrict__ psink,
    const float* __restrict__ v_nulls, unsigned short* __restrict__ ctx) {
    __shared__ float hbuf[16 * 256];
    __shared__ unsigned short hbf[16 * 256];
    __shared__ float scl[16];
    int tt = blockIdx.x;
    int h  = tt >> 6;
    int ts = (tt & 63) * 16;
    int lane = threadIdx.x & 31;
    int w    = threadIdx.x >> 5;
    int mh   = (lane >> 4) & 1;
    int nn   = lane & 15;

    const __bf16* Ap = marker + ((size_t)h * T_SEQ + ts) * DH;
    v16bf a0 = load_a_frag_bf(Ap, DH, lane);
    v16bf a1 = load_a_frag_bf(Ap + 32, DH, lane);
#pragma unroll
    for (int sub = 0; sub < 2; ++sub) {          // 8 waves x 2 = 16 N-tiles of fc
        int nt = w * 2 + sub;
        v16bf b0 = load_b_frag_bf(fct + (size_t)nt * 16 * DH, DH, lane);
        v16bf b1 = load_b_frag_bf(fct + (size_t)nt * 16 * DH + 32, DH, lane);
        v8f c = {};
        c = wmma_bf16(a0, b0, c);
        c = wmma_bf16(a1, b1, c);
#pragma unroll
        for (int v = 0; v < 8; ++v) {
            int n = nt * 16 + nn;
            float val = c[v] + fc_b[n];
            float v2  = val * val;
            hbuf[(v + 8 * mh) * 256 + n] = v2 + 0.75f * v2 * val;
        }
    }
    __syncthreads();
    if (threadIdx.x < 16) {
        const float* hr = hbuf + threadIdx.x * 256;
        float ss = 0.f;
        for (int j = 0; j < 256; ++j) { float x = hr[j]; ss += x * x; }
        scl[threadIdx.x] = rsqrtf(ss * (1.0f / 256.0f) + EPSF);
    }
    __syncthreads();
    {   // swish + bf16 pack: 16 consecutive elements per thread (one row seg)
        float s = scl[(threadIdx.x * 16) >> 8];
        unsigned int* hb = (unsigned int*)hbf + threadIdx.x * 8;
        float v[16];
#pragma unroll
        for (int j = 0; j < 16; ++j) {
            float x = hbuf[threadIdx.x * 16 + j] * s;
            v[j] = x / (1.0f + __expf(-1.8137993642342178f * x));
        }
#pragma unroll
        for (int j = 0; j < 8; ++j) hb[j] = pack2bf(v[2 * j], v[2 * j + 1]);
    }
    __syncthreads();
    if (w < 4) {                                  // 4 N-tiles of proj (N=64)
        v8f c = {};
#pragma unroll
        for (int kc = 0; kc < 8; ++kc) {
            v16bf a = load_a_frag_bf((const __bf16*)hbf + kc * 32, 256, lane);
            v16bf b = load_b_frag_bf(projt + (size_t)w * 16 * 256 + kc * 32, 256, lane);
            c = wmma_bf16(a, b, c);
        }
        int n = h / NH, hh = h % NH;
#pragma unroll
        for (int v = 0; v < 8; ++v) {
            int m = v + 8 * mh;
            int t = ts + m;
            int d = w * 16 + nn;
            float val = c[v] + proj_b[d] +
                        psink[(size_t)h * T_SEQ + t] *
                        v_nulls[(size_t)n * C_DIM + hh * DH + d];
            ctx[((size_t)n * T_SEQ + t) * C_DIM + hh * DH + d] = f2bf(val);
        }
    }
}

// ----------------- WO projection, 16x64 per-wave, branch mean fused into accum
__global__ __launch_bounds__(256) void out_proj_kernel(
    const __bf16* __restrict__ ctx, const __bf16* __restrict__ wot,
    const float* __restrict__ WO_b, float* __restrict__ out) {
    int lane = threadIdx.x & 31;
    int wid  = blockIdx.x * 8 + (threadIdx.x >> 5);
    int nblk = C_DIM >> 6;                    // 12 column blocks of 64
    int tm   = wid / nblk;
    int tn   = wid % nblk;
    v8f cacc[4] = {};
    for (int nb = 0; nb < NBR; ++nb) {
        const __bf16* Ap = ctx + ((size_t)nb * T_SEQ + tm * 16) * C_DIM;
        const __bf16* Wp = wot + (size_t)nb * C_DIM * C_DIM + (size_t)tn * 64 * C_DIM;
        for (int k0 = 0; k0 < C_DIM; k0 += 32) {
            __builtin_prefetch(Ap + k0 + 64, 0, 3);
            v16bf a = load_a_frag_bf(Ap + k0, C_DIM, lane);
#pragma unroll
            for (int j = 0; j < 4; ++j) {
                v16bf b = load_b_frag_bf(Wp + (size_t)j * 16 * C_DIM + k0, C_DIM, lane);
                cacc[j] = wmma_bf16(a, b, cacc[j]);
            }
        }
    }
    int mh = (lane >> 4) & 1;
    int nn = lane & 15;
#pragma unroll
    for (int j = 0; j < 4; ++j) {
        int col  = tn * 64 + j * 16 + nn;
        float bs = 0.25f * (WO_b[col] + WO_b[C_DIM + col] +
                            WO_b[2 * C_DIM + col] + WO_b[3 * C_DIM + col]);
#pragma unroll
        for (int v = 0; v < 8; ++v) {
            int m = tm * 16 + v + 8 * mh;
            out[(size_t)m * C_DIM + col] = 0.25f * cacc[j][v] + bs;
        }
    }
}

// ----------------------------------------------------------------- launcher
extern "C" void kernel_launch(void* const* d_in, const int* in_sizes, int n_in,
                              void* d_out, int out_size, void* d_ws, size_t ws_size,
                              hipStream_t stream) {
    (void)in_sizes; (void)n_in; (void)out_size; (void)ws_size;
    const float* A       = (const float*)d_in[0];
    const float* X       = (const float*)d_in[1];
    const float* WK_w    = (const float*)d_in[2];
    const float* WK_b    = (const float*)d_in[3];
    const float* WQ_w    = (const float*)d_in[4];
    const float* WQ_b    = (const float*)d_in[5];
    const float* wedge_A = (const float*)d_in[6];
    const float* wedge_b = (const float*)d_in[7];
    const float* sink    = (const float*)d_in[8];
    const float* v_nulls = (const float*)d_in[9];
    const float* fc_w    = (const float*)d_in[10];
    const float* fc_b    = (const float*)d_in[11];
    const float* proj_w  = (const float*)d_in[12];
    const float* proj_b  = (const float*)d_in[13];
    const float* WO      = (const float*)d_in[14];
    const float* WO_b    = (const float*)d_in[15];
    float* out = (float*)d_out;

    char* cur = (char*)d_ws;
    auto alloc = [&](size_t bytes) {
        char* p = cur;
        cur += (bytes + 255) & ~(size_t)255;
        return p;
    };
    unsigned short* abf   = (unsigned short*)alloc((size_t)T_SEQ * C_DIM * 2);
    unsigned short* xbf   = (unsigned short*)alloc((size_t)T_SEQ * C_DIM * 2);
    unsigned short* wqt   = (unsigned short*)alloc((size_t)3072 * C_DIM * 2);
    unsigned short* wkt   = (unsigned short*)alloc((size_t)C_DIM * C_DIM * 2);
    unsigned short* fct   = (unsigned short*)alloc((size_t)256 * DH * 2);
    unsigned short* prjt  = (unsigned short*)alloc((size_t)DH * 256 * 2);
    unsigned short* wot   = (unsigned short*)alloc((size_t)NBR * C_DIM * C_DIM * 2);
    float*          q_lin = (float*)alloc((size_t)T_SEQ * 3072 * 4);
    float*          k_lin = (float*)alloc((size_t)T_SEQ * C_DIM * 4);
    unsigned short* qrbf  = (unsigned short*)alloc((size_t)HT * T_SEQ * DH * 2);
    unsigned short* krbf  = (unsigned short*)alloc((size_t)HT * T_SEQ * DH * 2);
    float*          kbash = (float*)alloc((size_t)NH * T_SEQ * DH * 4);
    unsigned short* mkbf  = (unsigned short*)alloc((size_t)HT * T_SEQ * DH * 2);
    float*          psink = (float*)alloc((size_t)HT * T_SEQ * 4);
    unsigned short* ctxbf = (unsigned short*)alloc((size_t)NBR * T_SEQ * C_DIM * 2);

    // prologue: bf16 conversions / weight transposes
    {
        int n2 = (T_SEQ * C_DIM) / 2;
        cvt_bf16_kernel<<<(n2 + 255) / 256, 256, 0, stream>>>(A, (unsigned int*)abf, n2);
        cvt_bf16_kernel<<<(n2 + 255) / 256, 256, 0, stream>>>(X, (unsigned int*)xbf, n2);
    }
    transpose_bf16_kernel<<<dim3(C_DIM / 32, 3072 / 32, 1), 256, 0, stream>>>(WQ_w, wqt, C_DIM, 3072);
    transpose_bf16_kernel<<<dim3(C_DIM / 32, C_DIM / 32, 1), 256, 0, stream>>>(WK_w, wkt, C_DIM, C_DIM);
    transpose_bf16_kernel<<<dim3(DH / 32, 256 / 32, 1), 256, 0, stream>>>(fc_w, fct, DH, 256);
    transpose_bf16_kernel<<<dim3(256 / 32, DH / 32, 1), 256, 0, stream>>>(proj_w, prjt, 256, DH);
    transpose_bf16_kernel<<<dim3(C_DIM / 32, C_DIM / 32, NBR), 256, 0, stream>>>(WO, wot, C_DIM, C_DIM);

    // projections (16x64 tiles per wave)
    {
        int tiles = (T_SEQ / 16) * ((NBR * C_DIM) / 64);
        gemm_bias_kernel<<<(tiles + 7) / 8, 256, 0, stream>>>(
            (const __bf16*)abf, (const __bf16*)wqt, WQ_b, q_lin, T_SEQ, C_DIM, NBR * C_DIM);
    }
    {
        int tiles = (T_SEQ / 16) * (C_DIM / 64);
        gemm_bias_kernel<<<(tiles + 7) / 8, 256, 0, stream>>>(
            (const __bf16*)xbf, (const __bf16*)wkt, WK_b, k_lin, T_SEQ, C_DIM, C_DIM);
    }
    dim3 gprep(HT, T_SEQ / 256);
    qprep_kernel<<<gprep, 256, 0, stream>>>(q_lin, wedge_A, wedge_b, qrbf);
    kprep_kernel<<<gprep, 256, 0, stream>>>(k_lin, wedge_A, wedge_b, krbf, kbash);

    dim3 gattn(T_SEQ / 16, HT);
    attn_topk_kernel<<<gattn, 512, 0, stream>>>(
        (const __bf16*)qrbf, (const __bf16*)krbf, kbash, sink, mkbf, psink);

    mlp_ctx_kernel<<<HT * (T_SEQ / 16), 256, 0, stream>>>(
        (const __bf16*)mkbf, (const __bf16*)fct, fc_b, (const __bf16*)prjt,
        proj_b, psink, v_nulls, ctxbf);

    out_proj_kernel<<<((T_SEQ / 16) * (C_DIM / 64)) / 8, 256, 0, stream>>>(
        (const __bf16*)ctxbf, (const __bf16*)wot, WO_b, out);
}
